// EfficientMultiHeadAttention_45054206935229
// MI455X (gfx1250) — matrix operand order repaired and checked by reading, not compiled, not run on hardware
//
#include <hip/hip_runtime.h>

// ---------------------------------------------------------------------------
// Problem constants (match reference: B=2, T=2048, D=2048, H=16, K=128)
// ---------------------------------------------------------------------------
#define BSZ 2
#define SEQ 2048
#define DIM 2048          // model dim == H*K
#define NH  16
#define HK  128           // head size
#define MROWS (BSZ * SEQ) // 4096 flattened rows

typedef __bf16 bf16;
typedef __attribute__((ext_vector_type(16))) __bf16 v16bf;
typedef __attribute__((ext_vector_type(8)))  __bf16 v8bf;
typedef __attribute__((ext_vector_type(8)))  float  v8f;
typedef __attribute__((ext_vector_type(4)))  float  v4f;

union v16u {
    v16bf v;
    struct { v8bf lo, hi; } s;
};

// f32 -> bf16 round-to-nearest-even
__device__ __forceinline__ bf16 f2bf(float f) {
    unsigned u = __builtin_bit_cast(unsigned, f);
    u += 0x7FFFu + ((u >> 16) & 1u);
    unsigned short h = (unsigned short)(u >> 16);
    return __builtin_bit_cast(bf16, h);
}

// ---------------------------------------------------------------------------
// WMMA fragment loaders (layouts per CDNA5 ISA 7.12.2, wave32)
// A (16x32, bf16): lane l holds row m=l%16; elem e -> K=(e/8)*16+(l/16)*8+(e%8)
// B (32x16, bf16): lane l holds col n=l%16; elem e -> K=(e/8)*16+(l/16)*8+(e%8)
// C/D (16x16 f32): lane l holds col n=l%16; VGPR r -> row m=(l/16)*8+r
// ---------------------------------------------------------------------------
__device__ __forceinline__ v16bf load_a_bf16(const bf16* A, int lda, int lane) {
    const bf16* p0 = A + (lane & 15) * lda + ((lane >> 4) << 3);
    v16u u;
    u.s.lo = *(const v8bf*)p0;
    u.s.hi = *(const v8bf*)(p0 + 16);
    return u.v;
}

// B fragment from an N x Kdim row-major matrix (logical B transposed in
// memory): per lane two 16-byte loads.
__device__ __forceinline__ v16bf load_b_nk(const bf16* Bt, int ldb, int lane) {
    const bf16* p = Bt + (lane & 15) * ldb + ((lane >> 4) << 3);
    v16u u;
    u.s.lo = *(const v8bf*)p;
    u.s.hi = *(const v8bf*)(p + 16);
    return u.v;
}

__device__ __forceinline__ v8f wmma_bf16(v16bf a, v16bf b, v8f c) {
    return __builtin_amdgcn_wmma_f32_16x16x32_bf16(false, a, false, b,
                                                   (short)0, c, false, false);
}

// ---------------------------------------------------------------------------
// Kernel 0: elementwise f32 -> bf16 convert (vector loads/stores)
// ---------------------------------------------------------------------------
__global__ __launch_bounds__(256) void cvt_kernel(const float* __restrict__ X,
                                                  bf16* __restrict__ Y) {
    int i = (blockIdx.x * 256 + threadIdx.x) * 8;
    v4f x0 = *(const v4f*)(X + i);
    v4f x1 = *(const v4f*)(X + i + 4);
    v8bf y;
#pragma unroll
    for (int j = 0; j < 4; ++j) {
        y[j]     = f2bf(x0[j]);
        y[4 + j] = f2bf(x1[j]);
    }
    *(v8bf*)(Y + i) = y;
}

// ---------------------------------------------------------------------------
// Kernel 1: weight transpose + f32->bf16  (Wt[n][k] = W[k][n])
// ---------------------------------------------------------------------------
__global__ __launch_bounds__(256) void wtrans_kernel(const float* __restrict__ W,
                                                     bf16* __restrict__ Wt) {
    __shared__ float tile[32][33];
    int bx = blockIdx.x * 32, by = blockIdx.y * 32;
    int tx = threadIdx.x & 31;
    int ty = (threadIdx.x >> 5) * 4;
#pragma unroll
    for (int i = 0; i < 4; ++i)
        tile[ty + i][tx] = W[(by + ty + i) * DIM + bx + tx];
    __syncthreads();
#pragma unroll
    for (int i = 0; i < 4; ++i)
        Wt[(bx + ty + i) * DIM + by + tx] = f2bf(tile[tx][ty + i]);
}

// ---------------------------------------------------------------------------
// Kernel 2: unified WMMA GEMM  C = A(bf16, M x DIM) * Wt^T + bias
//   mode 0: bf16 out, row-major           (Q, K projections)
//   mode 1: bf16 out, V transposed        out_vt[((b*NH+h)*HK+d)][t]
//   mode 2: f32 out, row-major            (final output projection)
// Block: 256 thr = 8 waves in 2x4 grid; block tile 128x256; wave tile 64x64.
// Inner loop: pure b128 loads + 16 WMMAs per K-step; unconditional prefetch
// (speculative, contiguous workspace => always a valid device address).
// ---------------------------------------------------------------------------
__global__ __launch_bounds__(256) void gemm_bf16(const bf16* __restrict__ A,
                                                 const bf16* __restrict__ Wt,
                                                 const float* __restrict__ bias,
                                                 bf16* __restrict__ outb,
                                                 float* __restrict__ outf,
                                                 int mode) {
    int lane = threadIdx.x & 31;
    int w = threadIdx.x >> 5;
    int wm = w >> 2, wn = w & 3;
    int mBase = blockIdx.y * 128 + wm * 64;
    int nBase = blockIdx.x * 256 + wn * 64;

    v8f c[4][4] = {};
    for (int kb = 0; kb < DIM; kb += 32) {
        __builtin_prefetch(Wt + (nBase + (lane & 15)) * DIM + kb + 32, 0, 1);
        __builtin_prefetch(A + (mBase + (lane & 15)) * DIM + kb + 32, 0, 1);
        v16bf bfr[4];
#pragma unroll
        for (int nt = 0; nt < 4; ++nt)
            bfr[nt] = load_b_nk(Wt + (nBase + nt * 16) * DIM + kb, DIM, lane);
#pragma unroll
        for (int mt = 0; mt < 4; ++mt) {
            v16bf a = load_a_bf16(A + (mBase + mt * 16) * DIM + kb, DIM, lane);
#pragma unroll
            for (int nt = 0; nt < 4; ++nt)
                c[mt][nt] = wmma_bf16(a, bfr[nt], c[mt][nt]);
        }
    }

    int n = lane & 15;
    int m0 = (lane >> 4) << 3;
#pragma unroll
    for (int mt = 0; mt < 4; ++mt) {
#pragma unroll
        for (int nt = 0; nt < 4; ++nt) {
            int col = nBase + nt * 16 + n;
            float bv = bias[col];
            if (mode == 0) {
#pragma unroll
                for (int r = 0; r < 8; ++r) {
                    int row = mBase + mt * 16 + m0 + r;
                    outb[row * DIM + col] = f2bf(c[mt][nt][r] + bv);
                }
            } else if (mode == 1) {
                int h = col >> 7, d = col & (HK - 1);
#pragma unroll
                for (int r = 0; r < 8; ++r) {
                    int row = mBase + mt * 16 + m0 + r;   // flat b*SEQ + t
                    int bb = row >> 11, t = row & (SEQ - 1);
                    outb[(((bb * NH + h) << 7) + d) * SEQ + t] =
                        f2bf(c[mt][nt][r] + bv);
                }
            } else {
#pragma unroll
                for (int r = 0; r < 8; ++r) {
                    int row = mBase + mt * 16 + m0 + r;
                    outf[(size_t)row * DIM + col] = c[mt][nt][r] + bv;
                }
            }
        }
    }
}

// ---------------------------------------------------------------------------
// Kernel 3: flash attention.  One wave = 16 queries of one (b,h).
// Key tiles of 64 (amortizes softmax shuffles / O-rescale over 2x keys);
// online softmax; P re-laid out C->A via per-wave LDS tile.
// ---------------------------------------------------------------------------
__global__ __launch_bounds__(256) void attn_kernel(const bf16* __restrict__ qws,
                                                   const bf16* __restrict__ kws,
                                                   const bf16* __restrict__ vt,
                                                   bf16* __restrict__ attnws) {
    __shared__ bf16 plds[8][16 * 64];   // per-wave 16x64 P tile (bf16), 16 KiB
    int lane = threadIdx.x & 31;
    int w = threadIdx.x >> 5;
    int b = blockIdx.z, h = blockIdx.y;
    int q0 = blockIdx.x * 128 + w * 16;
    const float scale = 0.08838834764831845f;  // 1/sqrt(128)

    const bf16* Qb = qws + (size_t)(b * SEQ + q0) * DIM + h * HK;
    const bf16* Kb = kws + (size_t)(b * SEQ) * DIM + h * HK;
    const bf16* Vb = vt + (size_t)((b * NH + h) * HK) * SEQ;
    bf16* myp = plds[w];

    v16bf aq[4];
#pragma unroll
    for (int kk = 0; kk < 4; ++kk) aq[kk] = load_a_bf16(Qb + kk * 32, DIM, lane);

    v8f o[8] = {};
    float mrow[8], lrow[8];
#pragma unroll
    for (int r = 0; r < 8; ++r) { mrow[r] = -__builtin_inff(); lrow[r] = 0.f; }

    int m0 = (lane >> 4) << 3;
    int n = lane & 15;

    for (int j = 0; j < SEQ; j += 64) {
        // ---- S = (Q Kt) tile: 16 queries x 64 keys, 16 WMMAs ------------
        v8f s[4] = {};
#pragma unroll
        for (int kk = 0; kk < 4; ++kk) {
#pragma unroll
            for (int q = 0; q < 4; ++q) {
                v16bf bk = load_b_nk(Kb + (j + q * 16) * DIM + kk * 32, DIM, lane);
                s[q] = wmma_bf16(aq[kk], bk, s[q]);
            }
        }
        // ---- online softmax row stats (rows live in 16-lane halves) -----
        float alpha[8];
#pragma unroll
        for (int r = 0; r < 8; ++r) {
            float v0 = s[0][r] * scale, v1 = s[1][r] * scale;
            float v2 = s[2][r] * scale, v3 = s[3][r] * scale;
            float mx = fmaxf(fmaxf(v0, v1), fmaxf(v2, v3));
            mx = fmaxf(mx, __shfl_xor(mx, 1));
            mx = fmaxf(mx, __shfl_xor(mx, 2));
            mx = fmaxf(mx, __shfl_xor(mx, 4));
            mx = fmaxf(mx, __shfl_xor(mx, 8));
            float mnew = fmaxf(mrow[r], mx);
            alpha[r] = __expf(mrow[r] - mnew);
            float p0 = __expf(v0 - mnew);
            float p1 = __expf(v1 - mnew);
            float p2 = __expf(v2 - mnew);
            float p3 = __expf(v3 - mnew);
            float rs = (p0 + p1) + (p2 + p3);
            rs += __shfl_xor(rs, 1);
            rs += __shfl_xor(rs, 2);
            rs += __shfl_xor(rs, 4);
            rs += __shfl_xor(rs, 8);
            lrow[r] = lrow[r] * alpha[r] + rs;
            mrow[r] = mnew;
            int m = m0 + r;
            myp[m * 64 + n]      = f2bf(p0);   // key cols  0..15
            myp[m * 64 + 16 + n] = f2bf(p1);   // key cols 16..31
            myp[m * 64 + 32 + n] = f2bf(p2);   // key cols 32..47
            myp[m * 64 + 48 + n] = f2bf(p3);   // key cols 48..63
        }
        // ---- rescale accumulator ----------------------------------------
#pragma unroll
        for (int nt = 0; nt < 8; ++nt)
#pragma unroll
            for (int r = 0; r < 8; ++r) o[nt][r] *= alpha[r];

        // LDS within a wave is in-order; just stop compiler reordering.
        asm volatile("" ::: "memory");

        // ---- reload P as two A fragments (C-layout -> A-layout) ---------
        const bf16* pr = myp + (lane & 15) * 64 + ((lane >> 4) << 3);
        v16u u0, u1;
        u0.s.lo = *(const v8bf*)pr;
        u0.s.hi = *(const v8bf*)(pr + 16);
        u1.s.lo = *(const v8bf*)(pr + 32);
        u1.s.hi = *(const v8bf*)(pr + 48);
        v16bf ap0 = u0.v, ap1 = u1.v;

        // ---- O += P * V tile (64 keys x 128 head dims), 16 WMMAs --------
#pragma unroll
        for (int nt = 0; nt < 8; ++nt) {
            v16bf bv0 = load_b_nk(Vb + (nt * 16) * SEQ + j,      SEQ, lane);
            v16bf bv1 = load_b_nk(Vb + (nt * 16) * SEQ + j + 32, SEQ, lane);
            o[nt] = wmma_bf16(ap0, bv0, o[nt]);
            o[nt] = wmma_bf16(ap1, bv1, o[nt]);
        }
    }

    // ---- normalize and store attn[b][t][h*HK + d] as bf16 ---------------
#pragma unroll
    for (int r = 0; r < 8; ++r) lrow[r] = 1.0f / lrow[r];
#pragma unroll
    for (int nt = 0; nt < 8; ++nt) {
#pragma unroll
        for (int r = 0; r < 8; ++r) {
            size_t row = (size_t)(b * SEQ + q0 + m0 + r);
            attnws[row * DIM + h * HK + nt * 16 + n] = f2bf(o[nt][r] * lrow[r]);
        }
    }
}

// ---------------------------------------------------------------------------
// Host launch
// ---------------------------------------------------------------------------
extern "C" void kernel_launch(void* const* d_in, const int* in_sizes, int n_in,
                              void* d_out, int out_size, void* d_ws, size_t ws_size,
                              hipStream_t stream) {
    (void)in_sizes; (void)n_in; (void)out_size; (void)ws_size;
    const float* query = (const float*)d_in[0];
    const float* key_  = (const float*)d_in[1];
    const float* value = (const float*)d_in[2];
    const float* wq = (const float*)d_in[3];
    const float* bq = (const float*)d_in[4];
    const float* wk = (const float*)d_in[5];
    const float* bk = (const float*)d_in[6];
    const float* wv = (const float*)d_in[7];
    const float* bv = (const float*)d_in[8];
    const float* wo = (const float*)d_in[9];
    const float* bo = (const float*)d_in[10];

    // Workspace layout (128 MiB total; attnws aliases xq which is dead by then)
    const size_t WSZ = (size_t)DIM * DIM * sizeof(bf16);       // 8 MiB
    const size_t ASZ = (size_t)MROWS * DIM * sizeof(bf16);     // 16 MiB
    char* ws = (char*)d_ws;
    bf16* wqT = (bf16*)(ws + 0 * WSZ);
    bf16* wkT = (bf16*)(ws + 1 * WSZ);
    bf16* wvT = (bf16*)(ws + 2 * WSZ);
    bf16* woT = (bf16*)(ws + 3 * WSZ);
    char* base = ws + 4 * WSZ;
    bf16* xq     = (bf16*)(base + 0 * ASZ);
    bf16* xk     = (bf16*)(base + 1 * ASZ);
    bf16* xv     = (bf16*)(base + 2 * ASZ);
    bf16* qws    = (bf16*)(base + 3 * ASZ);
    bf16* kws    = (bf16*)(base + 4 * ASZ);
    bf16* vtws   = (bf16*)(base + 5 * ASZ);
    bf16* attnws = xq;   // alias: query activations are dead after projection

    dim3 tb(256);

    // activations f32 -> bf16
    int cvtBlocks = (MROWS * DIM) / (256 * 8);   // 4096
    cvt_kernel<<<cvtBlocks, tb, 0, stream>>>(query, xq);
    cvt_kernel<<<cvtBlocks, tb, 0, stream>>>(key_, xk);
    cvt_kernel<<<cvtBlocks, tb, 0, stream>>>(value, xv);

    // weights: transpose + convert
    dim3 tg(DIM / 32, DIM / 32);                 // 64 x 64
    wtrans_kernel<<<tg, tb, 0, stream>>>(wq, wqT);
    wtrans_kernel<<<tg, tb, 0, stream>>>(wk, wkT);
    wtrans_kernel<<<tg, tb, 0, stream>>>(wv, wvT);
    wtrans_kernel<<<tg, tb, 0, stream>>>(wo, woT);

    // projections
    dim3 gg(DIM / 256, MROWS / 128);             // 8 x 32
    gemm_bf16<<<gg, tb, 0, stream>>>(xq, wqT, bq, qws, nullptr, 0);
    gemm_bf16<<<gg, tb, 0, stream>>>(xk, wkT, bk, kws, nullptr, 0);
    gemm_bf16<<<gg, tb, 0, stream>>>(xv, wvT, bv, vtws, nullptr, 1);

    // attention
    dim3 ga(SEQ / 128, NH, BSZ);                 // 16 x 16 x 2
    attn_kernel<<<ga, tb, 0, stream>>>(qws, kws, vtws, attnws);

    // output projection -> f32
    gemm_bf16<<<gg, tb, 0, stream>>>(attnws, woT, bo, nullptr, (float*)d_out, 2);
}
